// TransformerEncoderLayer_36137854828974
// MI455X (gfx1250) — matrix operand order repaired and checked by reading, not compiled, not run on hardware
//
#include <hip/hip_runtime.h>
#include <hip/hip_bf16.h>
#include <math.h>

// ---------------------------------------------------------------------------
// TransformerEncoderLayer for MI455X (gfx1250, wave32, WMMA, TDM/async-LDS).
// fp32 in/out, bf16 WMMA compute with fp32 accumulation.
// B=2 S=2048 E=1024 H=16 Dh=64 I=4096
// ---------------------------------------------------------------------------

typedef __bf16 bf16;
typedef __attribute__((ext_vector_type(16))) __bf16 v16bf;
typedef __attribute__((ext_vector_type(8)))  __bf16 v8bf;
typedef __attribute__((ext_vector_type(8)))  float  v8f;
typedef unsigned int  uint4_ev __attribute__((ext_vector_type(4)));
typedef int           int4_ev  __attribute__((ext_vector_type(4)));
typedef int           int8_ev  __attribute__((ext_vector_type(8)));
typedef int           v4i_vs   __attribute__((vector_size(16)));

constexpr int Bsz = 2;
constexpr int S   = 2048;
constexpr int E   = 1024;
constexpr int H   = 16;
constexpr int Dh  = 64;
constexpr int I   = 4096;
constexpr int M   = Bsz * S;           // 4096 rows
constexpr float EPS = 1e-5f;

// --------- CDNA5 data-movement feature detection (compile-safe) -----------
#if __has_builtin(__builtin_amdgcn_tensor_load_to_lds) && \
    __has_builtin(__builtin_amdgcn_s_wait_tensorcnt)
#define HAVE_TDM 1
#endif
#if __has_builtin(__builtin_amdgcn_global_load_async_to_lds_b128) && \
    __has_builtin(__builtin_amdgcn_s_wait_asynccnt)
#define HAVE_ASYNC 1
#endif

// ---------------- helpers ----------------

__device__ inline v8f zero8() {
  v8f z;
#pragma unroll
  for (int i = 0; i < 8; ++i) z[i] = 0.0f;
  return z;
}

__device__ inline v16bf ld_frag16(const bf16* p0, const bf16* p1) {
  v8bf a = *(const v8bf*)p0;
  v8bf b = *(const v8bf*)p1;
  return __builtin_shufflevector(a, b, 0,1,2,3,4,5,6,7,8,9,10,11,12,13,14,15);
}

__device__ inline v8f wmma_bf16(v16bf a, v16bf b, v8f c) {
  return __builtin_amdgcn_wmma_f32_16x16x32_bf16(
      false, a, false, b, (short)0, c, false, false);
}

#if defined(HAVE_TDM)
__device__ inline unsigned lds_byte_off(const void* p) {
  return (unsigned)(unsigned long long)
      (__attribute__((address_space(3))) const char*)p;
}
// 2-D tile load via Tensor Data Mover: tileW x tileH elements (bf16),
// source row stride srcStride (elements), packed contiguously into LDS.
__device__ inline void tdm_load_2d(const bf16* gsrc, unsigned ldsOff,
                                   unsigned tileW, unsigned tileH,
                                   unsigned srcStride) {
  unsigned long long ga = (unsigned long long)gsrc;
  unsigned td0 = srcStride;          // tensor_dim0 >= tileW (no clipping)
  unsigned td1 = tileH;              // tensor_dim1
  uint4_ev g0;
  g0[0] = 1u;                                            // count=1, user mode
  g0[1] = ldsOff;                                        // lds_addr
  g0[2] = (unsigned)(ga & 0xffffffffu);                  // global_addr lo
  g0[3] = (unsigned)((ga >> 32) & 0x01ffffffu) | (2u << 30);  // hi | type=2
  int8_ev g1;
  g1[0] = (int)(1u << 16);                               // data_size=2B, mask=0
  g1[1] = (int)((td0 & 0xffffu) << 16);                  // tensor_dim0[15:0]
  g1[2] = (int)((td0 >> 16) | ((td1 & 0xffffu) << 16));  // td0 hi | td1 lo
  g1[3] = (int)((td1 >> 16) | (tileW << 16));            // td1 hi | tile_dim0
  g1[4] = (int)(tileH & 0xffffu);                        // tile_dim1, tile_dim2=0
  g1[5] = (int)srcStride;                                // dim0_stride lo
  g1[6] = 0;                                             // dim0_stride hi
  g1[7] = 0;
  int4_ev gz = {0, 0, 0, 0};
#if defined(__clang_major__) && __clang_major__ >= 23
  int8_ev gz8 = {0, 0, 0, 0, 0, 0, 0, 0};
  __builtin_amdgcn_tensor_load_to_lds(g0, g1, gz, gz, gz8, 0);
#else
  __builtin_amdgcn_tensor_load_to_lds(g0, g1, gz, gz, 0);
#endif
}
#endif

#if defined(HAVE_ASYNC)
__device__ inline void async_cp16(const bf16* g, bf16* l) {
  // param 0: <4 x i32> in global (__device__) AS, param 1: LDS destination
  __builtin_amdgcn_global_load_async_to_lds_b128(
      (__attribute__((address_space(1))) v4i_vs*)(unsigned long long)g,
      (__attribute__((address_space(3))) v4i_vs*)l,
      0, 0);
}
#endif

// ---------------- f32 -> bf16 conversion ----------------

__global__ __launch_bounds__(256) void cvt_f32_bf16(const float* __restrict__ in,
                                                    bf16* __restrict__ out, int n) {
  int i = (blockIdx.x * 256 + threadIdx.x) * 4;
  if (i + 3 < n) {
    float4 v = *(const float4*)&in[i];
    out[i + 0] = (bf16)v.x;
    out[i + 1] = (bf16)v.y;
    out[i + 2] = (bf16)v.z;
    out[i + 3] = (bf16)v.w;
  } else {
    for (int j = i; j < n; ++j) out[j] = (bf16)in[j];
  }
}

// ---------------- LayerNorm (f32 -> bf16) ----------------

__global__ __launch_bounds__(256) void layernorm_bf16(const float* __restrict__ x,
                                                      const float* __restrict__ g,
                                                      const float* __restrict__ bb,
                                                      bf16* __restrict__ out) {
  int row = blockIdx.x;
  int tid = threadIdx.x;
  int wave = tid >> 5, lane = tid & 31;
  const float* xr = x + (size_t)row * E;
  int base = tid * 4;
  float4 xv = *(const float4*)&xr[base];
  float s  = xv.x + xv.y + xv.z + xv.w;
  float s2 = xv.x * xv.x + xv.y * xv.y + xv.z * xv.z + xv.w * xv.w;
#pragma unroll
  for (int off = 1; off < 32; off <<= 1) {
    s  += __shfl_xor(s,  off, 32);
    s2 += __shfl_xor(s2, off, 32);
  }
  __shared__ float red[16];
  if (lane == 0) { red[wave * 2] = s; red[wave * 2 + 1] = s2; }
  __syncthreads();
  s = 0.0f; s2 = 0.0f;
#pragma unroll
  for (int w = 0; w < 8; ++w) { s += red[w * 2]; s2 += red[w * 2 + 1]; }
  float mean = s * (1.0f / E);
  float var  = s2 * (1.0f / E) - mean * mean;
  float rstd = rsqrtf(var + EPS);
  float xs[4] = {xv.x, xv.y, xv.z, xv.w};
#pragma unroll
  for (int j = 0; j < 4; ++j) {
    float vv = (xs[j] - mean) * rstd * g[base + j] + bb[base + j];
    out[(size_t)row * E + base + j] = (bf16)vv;
  }
}

// ---------------- generic WMMA GEMM ----------------
// C[M,N] = A[M,K](bf16) x Bw[K,N](bf16) + bias.
// Block tile 256x64, BK=32, double-buffered LDS, 8 waves, each wave = 32x64.
// MODE 0: store bf16 (optional exact GELU). MODE 1: store f32 with residual.

constexpr int BM = 256, BN = 64, BK = 32;

template <int MODE, bool GELU>
__global__ __launch_bounds__(256) void gemm_bf16_wmma(
    const bf16* __restrict__ A, int lda,
    const bf16* __restrict__ Bw, int ldb, long strideB,
    const float* __restrict__ bias, long strideBias,
    float* __restrict__ Cf, bf16* __restrict__ Cb, int ldc, long strideC,
    const float* __restrict__ resid, int K) {
  __shared__ alignas(16) bf16 As[2][BM][BK];
  __shared__ alignas(16) bf16 BsT[2][BN][BK];   // transposed: [n][k]

  int tid  = threadIdx.x;
  int wave = tid >> 5, lane = tid & 31;
  int m0 = blockIdx.y * BM, n0 = blockIdx.x * BN;
  long z = blockIdx.z;
  const bf16*  Bz    = Bw + z * strideB;
  const float* biasZ = bias + z * strideBias;
  long coff = z * strideC;

  v8f acc[2][4];
#pragma unroll
  for (int mf = 0; mf < 2; ++mf)
#pragma unroll
    for (int nt = 0; nt < 4; ++nt) acc[mf][nt] = zero8();

  int wm  = wave * 32;
  int l15 = lane & 15;
  int kbA = (lane < 16) ? 0 : 8;
  int kbB = (lane < 16) ? 0 : 8;

  // ---- tile staging ----
  auto stageA = [&](int buf, int k0) {
#if defined(HAVE_TDM)
    if (wave == 0)
      tdm_load_2d(&A[(size_t)m0 * lda + k0], lds_byte_off(&As[buf][0][0]),
                  BK, BM, (unsigned)lda);
#elif defined(HAVE_ASYNC)
    int c0 = tid * 4;
#pragma unroll
    for (int i = 0; i < 4; ++i) {
      int c = c0 + i;
      int row = c >> 2, col = (c & 3) * 8;
      async_cp16(&A[(size_t)(m0 + row) * lda + k0 + col], &As[buf][row][col]);
    }
#else
    int c0 = tid * 4;
#pragma unroll
    for (int i = 0; i < 4; ++i) {
      int c = c0 + i;
      int row = c >> 2, col = (c & 3) * 8;
      *(v8bf*)&As[buf][row][col] =
          *(const v8bf*)&A[(size_t)(m0 + row) * lda + k0 + col];
    }
#endif
  };
  auto stageB = [&](int buf, int k0) {
    int kr  = tid & 31;
    int nc0 = (tid >> 5) * 8;
    v8bf tmp = *(const v8bf*)&Bz[(size_t)(k0 + kr) * ldb + n0 + nc0];
#pragma unroll
    for (int j = 0; j < 8; ++j) BsT[buf][nc0 + j][kr] = tmp[j];
  };
  auto waitStage = [&]() {
#if defined(HAVE_TDM)
    if (wave == 0) __builtin_amdgcn_s_wait_tensorcnt(0);
#elif defined(HAVE_ASYNC)
    __builtin_amdgcn_s_wait_asynccnt(0);
#endif
  };

  // prologue: fill buffer 0
  stageA(0, 0);
  stageB(0, 0);

  int iters = K / BK;
  for (int i = 0; i < iters; ++i) {
    int cur = i & 1;
    waitStage();
    __syncthreads();          // staging of cur visible; compute of i-1 done
    if (i + 1 < iters) {      // prefetch next tile into the other buffer
      stageA(cur ^ 1, (i + 1) * BK);
      stageB(cur ^ 1, (i + 1) * BK);
    }
    v16bf af0 = ld_frag16(&As[cur][wm + l15][kbA],      &As[cur][wm + l15][kbA + 16]);
    v16bf af1 = ld_frag16(&As[cur][wm + 16 + l15][kbA], &As[cur][wm + 16 + l15][kbA + 16]);
#pragma unroll
    for (int nt = 0; nt < 4; ++nt) {
      int n = nt * 16 + l15;
      v16bf bfrag = ld_frag16(&BsT[cur][n][kbB], &BsT[cur][n][kbB + 16]);
      acc[0][nt] = wmma_bf16(af0, bfrag, acc[0][nt]);
      acc[1][nt] = wmma_bf16(af1, bfrag, acc[1][nt]);
    }
  }

  // epilogue
#pragma unroll
  for (int mf = 0; mf < 2; ++mf) {
#pragma unroll
    for (int nt = 0; nt < 4; ++nt) {
#pragma unroll
      for (int r = 0; r < 8; ++r) {
        int grow = m0 + wm + mf * 16 + r + ((lane & 16) ? 8 : 0);
        int gcol = n0 + nt * 16 + l15;
        float vacc = acc[mf][nt][r] + biasZ[gcol];
        size_t idx = (size_t)grow * ldc + coff + gcol;
        if (MODE == 0) {
          float vv = vacc;
          if (GELU) vv = 0.5f * vv * (1.0f + erff(vv * 0.70710678118654752f));
          Cb[idx] = (bf16)vv;
        } else {
          Cf[idx] = resid[idx] + vacc;
        }
      }
    }
  }
}

// ---------------- flash attention ----------------
// q,k,v,o: bf16 [B, S, H*Dh]; block = (b,h) x 128 query rows, 8 waves.

__global__ __launch_bounds__(256) void flash_attn_wmma(
    const bf16* __restrict__ q, const bf16* __restrict__ k,
    const bf16* __restrict__ v, bf16* __restrict__ o) {
  __shared__ alignas(16) bf16 Ks[32][64];       // [t][d]
  __shared__ alignas(16) bf16 VsT[64][32];      // [d][t]
  __shared__ alignas(16) bf16 Ps[8][16][32];    // per-wave P scratch

  int tid  = threadIdx.x;
  int wave = tid >> 5, lane = tid & 31;
  int bh = blockIdx.y;
  int b = bh / H, h = bh % H;
  int m0 = blockIdx.x * 128;

  int mrow = m0 + wave * 16 + (lane & 15);
  size_t qrow = ((size_t)b * S + mrow) * E + (size_t)h * Dh;
  int kbA = (lane < 16) ? 0 : 8;
  int kbB = (lane < 16) ? 0 : 8;

  v16bf qf0 = ld_frag16(&q[qrow + kbA],      &q[qrow + kbA + 16]);
  v16bf qf1 = ld_frag16(&q[qrow + 32 + kbA], &q[qrow + 32 + kbA + 16]);

  float mi[8], li[8];
  v8f Oacc[4];
#pragma unroll
  for (int r = 0; r < 8; ++r) { mi[r] = -1e30f; li[r] = 0.0f; }
#pragma unroll
  for (int dt = 0; dt < 4; ++dt) Oacc[dt] = zero8();

  int tr = tid & 31, d0 = (tid >> 5) * 8;
  size_t kvbase = (size_t)b * S * E + (size_t)h * Dh;

  for (int t0 = 0; t0 < S; t0 += 32) {
    __syncthreads();
    {
      size_t g = kvbase + (size_t)(t0 + tr) * E + d0;
      // K tile: TDM / async / sync
#if defined(HAVE_TDM)
      if (wave == 0)
        tdm_load_2d(&k[kvbase + (size_t)t0 * E], lds_byte_off(&Ks[0][0]),
                    64, 32, (unsigned)E);
#elif defined(HAVE_ASYNC)
      async_cp16(&k[g], &Ks[tr][d0]);
#else
      *(v8bf*)&Ks[tr][d0] = *(const v8bf*)&k[g];
#endif
      // V tile transposed (scatter)
      v8bf tv = *(const v8bf*)&v[g];
#pragma unroll
      for (int j = 0; j < 8; ++j) VsT[d0 + j][tr] = tv[j];
    }
#if defined(HAVE_TDM)
    if (wave == 0) __builtin_amdgcn_s_wait_tensorcnt(0);
#elif defined(HAVE_ASYNC)
    __builtin_amdgcn_s_wait_asynccnt(0);
#endif
    __syncthreads();

    // scores: S[16x32] = Q(16x64) x K^T(64x32)
    v8f sc[2];
    sc[0] = zero8(); sc[1] = zero8();
#pragma unroll
    for (int nt = 0; nt < 2; ++nt) {
      int tl = nt * 16 + (lane & 15);
      v16bf b0 = ld_frag16(&Ks[tl][kbB],      &Ks[tl][kbB + 16]);
      v16bf b1 = ld_frag16(&Ks[tl][32 + kbB], &Ks[tl][32 + kbB + 16]);
      sc[nt] = wmma_bf16(qf0, b0, sc[nt]);
      sc[nt] = wmma_bf16(qf1, b1, sc[nt]);
    }

    // online softmax
#pragma unroll
    for (int r = 0; r < 8; ++r) {
      float s0 = sc[0][r] * 0.125f;   // 1/sqrt(64)
      float s1 = sc[1][r] * 0.125f;
      float mx = fmaxf(s0, s1);
#pragma unroll
      for (int off = 1; off < 16; off <<= 1) mx = fmaxf(mx, __shfl_xor(mx, off, 32));
      float mnew = fmaxf(mi[r], mx);
      float corr = __expf(mi[r] - mnew);
      float p0 = __expf(s0 - mnew);
      float p1 = __expf(s1 - mnew);
      float ps = p0 + p1;
#pragma unroll
      for (int off = 1; off < 16; off <<= 1) ps += __shfl_xor(ps, off, 32);
      li[r] = li[r] * corr + ps;
      mi[r] = mnew;
#pragma unroll
      for (int dt = 0; dt < 4; ++dt) Oacc[dt][r] *= corr;
      int prow = r + ((lane & 16) ? 8 : 0);
      Ps[wave][prow][(lane & 15)]      = (bf16)p0;
      Ps[wave][prow][(lane & 15) + 16] = (bf16)p1;
    }

    // O += P(16x32) x V(32x64)
    {
      int ml = lane & 15;
      v16bf pf = ld_frag16(&Ps[wave][ml][kbA], &Ps[wave][ml][kbA + 16]);
#pragma unroll
      for (int dt = 0; dt < 4; ++dt) {
        int n = dt * 16 + (lane & 15);
        v16bf vf = ld_frag16(&VsT[n][kbB], &VsT[n][kbB + 16]);
        Oacc[dt] = wmma_bf16(pf, vf, Oacc[dt]);
      }
    }
  }

  // normalize and store o in [B, S, H*Dh] bf16
#pragma unroll
  for (int dt = 0; dt < 4; ++dt) {
#pragma unroll
    for (int r = 0; r < 8; ++r) {
      int grow = m0 + wave * 16 + r + ((lane & 16) ? 8 : 0);
      int gcol = dt * 16 + (lane & 15);
      float val = Oacc[dt][r] / li[r];
      o[((size_t)b * S + grow) * E + (size_t)h * Dh + gcol] = (bf16)val;
    }
  }
}

// ---------------- launch ----------------

extern "C" void kernel_launch(void* const* d_in, const int* in_sizes, int n_in,
                              void* d_out, int out_size, void* d_ws, size_t ws_size,
                              hipStream_t stream) {
  const float* x     = (const float*)d_in[0];
  const float* Wq    = (const float*)d_in[1];
  const float* bq    = (const float*)d_in[2];
  const float* Wk    = (const float*)d_in[3];
  const float* bk    = (const float*)d_in[4];
  const float* Wv    = (const float*)d_in[5];
  const float* bv    = (const float*)d_in[6];
  const float* Wo    = (const float*)d_in[7];
  const float* bo    = (const float*)d_in[8];
  const float* ln1g  = (const float*)d_in[9];
  const float* ln1b  = (const float*)d_in[10];
  const float* ln2g  = (const float*)d_in[11];
  const float* ln2b  = (const float*)d_in[12];
  const float* W1    = (const float*)d_in[13];
  const float* b1    = (const float*)d_in[14];
  const float* W2    = (const float*)d_in[15];
  const float* b2    = (const float*)d_in[16];
  float* out = (float*)d_out;

  char* ws = (char*)d_ws;
  size_t off = 0;
  auto alloc = [&](size_t bytes) -> void* {
    void* p = ws + off;
    off += (bytes + 255) & ~(size_t)255;
    return p;
  };

  const int NWQKV = H * E * Dh;
  const int NWO   = E * E;
  const int NW1   = E * I;
  const int NW2   = I * E;

  bf16* wq = (bf16*)alloc((size_t)NWQKV * 2);
  bf16* wk = (bf16*)alloc((size_t)NWQKV * 2);
  bf16* wv = (bf16*)alloc((size_t)NWQKV * 2);
  bf16* wo = (bf16*)alloc((size_t)NWO * 2);
  bf16* w1 = (bf16*)alloc((size_t)NW1 * 2);
  bf16* w2 = (bf16*)alloc((size_t)NW2 * 2);
  bf16* h1 = (bf16*)alloc((size_t)M * E * 2);
  bf16* qb = (bf16*)alloc((size_t)M * E * 2);
  bf16* kb = (bf16*)alloc((size_t)M * E * 2);
  bf16* vb = (bf16*)alloc((size_t)M * E * 2);
  bf16* ob = (bf16*)alloc((size_t)M * E * 2);
  float* x2 = (float*)alloc((size_t)M * E * 4);
  bf16* h2 = (bf16*)alloc((size_t)M * E * 2);
  bf16* f1 = (bf16*)alloc((size_t)M * I * 2);
  (void)ws_size; (void)n_in; (void)in_sizes; (void)out_size;

  cvt_f32_bf16<<<(NWQKV / 4 + 255) / 256, 256, 0, stream>>>(Wq, wq, NWQKV);
  cvt_f32_bf16<<<(NWQKV / 4 + 255) / 256, 256, 0, stream>>>(Wk, wk, NWQKV);
  cvt_f32_bf16<<<(NWQKV / 4 + 255) / 256, 256, 0, stream>>>(Wv, wv, NWQKV);
  cvt_f32_bf16<<<(NWO   / 4 + 255) / 256, 256, 0, stream>>>(Wo, wo, NWO);
  cvt_f32_bf16<<<(NW1   / 4 + 255) / 256, 256, 0, stream>>>(W1, w1, NW1);
  cvt_f32_bf16<<<(NW2   / 4 + 255) / 256, 256, 0, stream>>>(W2, w2, NW2);

  layernorm_bf16<<<M, 256, 0, stream>>>(x, ln1g, ln1b, h1);

  {
    dim3 g(1, M / BM, H);
    gemm_bf16_wmma<0, false><<<g, 256, 0, stream>>>(
        h1, E, wq, Dh, (long)E * Dh, bq, (long)Dh,
        nullptr, qb, E, (long)Dh, nullptr, E);
    gemm_bf16_wmma<0, false><<<g, 256, 0, stream>>>(
        h1, E, wk, Dh, (long)E * Dh, bk, (long)Dh,
        nullptr, kb, E, (long)Dh, nullptr, E);
    gemm_bf16_wmma<0, false><<<g, 256, 0, stream>>>(
        h1, E, wv, Dh, (long)E * Dh, bv, (long)Dh,
        nullptr, vb, E, (long)Dh, nullptr, E);
  }

  flash_attn_wmma<<<dim3(S / 128, Bsz * H), 256, 0, stream>>>(qb, kb, vb, ob);

  gemm_bf16_wmma<1, false><<<dim3(E / BN, M / BM), 256, 0, stream>>>(
      ob, E, wo, E, 0L, bo, 0L, x2, nullptr, E, 0L, x, E);

  layernorm_bf16<<<M, 256, 0, stream>>>(x2, ln2g, ln2b, h2);

  gemm_bf16_wmma<0, true><<<dim3(I / BN, M / BM), 256, 0, stream>>>(
      h2, E, w1, I, 0L, b1, 0L, nullptr, f1, I, 0L, nullptr, E);

  gemm_bf16_wmma<1, false><<<dim3(E / BN, M / BM), 256, 0, stream>>>(
      f1, I, w2, E, 0L, b2, 0L, out, nullptr, E, 0L, x2, I);
}
